// GraphEGNNNet_68461778698588
// MI455X (gfx1250) — matrix operand order repaired and checked by reading, not compiled, not run on hardware
//
#include <hip/hip_runtime.h>
#include <hip/hip_bf16.h>

#define NN 10000
#define EE 160000
#define HD 128

typedef __attribute__((ext_vector_type(16))) _Float16 v16h;
typedef __attribute__((ext_vector_type(8)))  _Float16 v8h;
typedef __attribute__((ext_vector_type(8)))  float    v8f;

union FragA { v16h v; v8h h[2]; };

__device__ __forceinline__ float silu_f(float x){ return x / (1.f + __expf(-x)); }
__device__ __forceinline__ float sigm_f(float x){ return 1.f / (1.f + __expf(-x)); }

// ---------------- elementwise / prep kernels ----------------

__global__ __launch_bounds__(256) void k_fill0(float* p, int n){
  int i = blockIdx.x*256 + threadIdx.x;
  if (i < n) p[i] = 0.f;
}

// transpose fp32 [din,dout] -> f16 [dout,din] so B fragments are contiguous
__global__ __launch_bounds__(256) void k_wt_f16(const float* __restrict__ w,
                                                _Float16* __restrict__ wt, int din, int dout){
  int i = blockIdx.x*256 + threadIdx.x;
  if (i < din*dout){
    int k = i / dout, n = i % dout;
    wt[(size_t)n*din + k] = (_Float16)w[i];
  }
}

__global__ __launch_bounds__(256) void k_node_init(const int* __restrict__ h, const int* __restrict__ ind,
    const int* __restrict__ outd, const float* __restrict__ eh, const float* __restrict__ ei,
    const float* __restrict__ eo, float* __restrict__ hv, _Float16* __restrict__ hv16){
  int i = blockIdx.x*256 + threadIdx.x;
  if (i < NN*HD){
    int n = i>>7, j = i&127;
    float v = eh[h[n]*HD+j] + ei[ind[n]*HD+j] + eo[outd[n]*HD+j];
    hv[i] = v; hv16[i] = (_Float16)v;
  }
}

__global__ __launch_bounds__(256) void k_edge_pre(const int* __restrict__ e, const int* __restrict__ sp,
    const float* __restrict__ embe, const float* __restrict__ embsp,
    float* __restrict__ x, _Float16* __restrict__ pos16){
  int i = blockIdx.x*256 + threadIdx.x;
  if (i < EE*HD){
    int r = i>>7, j = i&127;
    float p = embsp[sp[r]*HD+j];
    x[i] = embe[e[r]*HD+j] + p;
    pos16[i] = (_Float16)p;
  }
}

__global__ __launch_bounds__(256) void k_deg(const int* __restrict__ dst, float* __restrict__ deg){
  int i = blockIdx.x*256 + threadIdx.x;
  if (i < EE) atomicAdd(&deg[dst[i]], 1.f);
}

__global__ __launch_bounds__(128) void k_scatter(const float* __restrict__ msg, const float* __restrict__ w,
    const int* __restrict__ dst, float* __restrict__ agg){
  int e = blockIdx.x; int j = threadIdx.x;
  atomicAdd(&agg[(size_t)dst[e]*HD + j], msg[(size_t)e*HD + j] * w[e]);
}

__global__ __launch_bounds__(256) void k_aggn(const float* __restrict__ agg, const float* __restrict__ deg,
                                              _Float16* __restrict__ aggn16){
  int i = blockIdx.x*256 + threadIdx.x;
  if (i < NN*HD){
    int n = i>>7;
    aggn16[i] = (_Float16)(agg[i] / (deg[n] + 1e-6f));
  }
}

// LayerNorm over H=128 per row; one row per block of 128 threads (4 waves, wave32)
__global__ __launch_bounds__(128) void k_ln(const float* __restrict__ x, const float* __restrict__ g,
    const float* __restrict__ b, float* __restrict__ y, _Float16* __restrict__ y16){
  int r = blockIdx.x, j = threadIdx.x;
  size_t o = (size_t)r*HD + j;
  float v = x[o];
  __shared__ float red[8];
  int wv = j>>5, ln = j&31;
  float s = v;
  #pragma unroll
  for (int t=16;t>0;t>>=1) s += __shfl_down(s, t);
  if (ln==0) red[wv] = s;
  __syncthreads();
  if (j==0) red[4] = (red[0]+red[1]+red[2]+red[3]) * (1.f/HD);
  __syncthreads();
  float m = red[4];
  float d = v - m;
  float q = d*d;
  #pragma unroll
  for (int t=16;t>0;t>>=1) q += __shfl_down(q, t);
  if (ln==0) red[wv] = q;
  __syncthreads();
  if (j==0) red[5] = rsqrtf((red[0]+red[1]+red[2]+red[3])*(1.f/HD) + 1e-5f);
  __syncthreads();
  float out = d * red[5] * g[j] + b[j];
  y[o] = out;
  if (y16) y16[o] = (_Float16)out;
}

// ---------------- fused 2-layer MLP GEMM (WMMA) ----------------
// Block: 128 threads = 4 waves; each wave computes a 16-row x 128-col tile.
// A rows gathered from up to 3 H-wide f16 segments (fused concat+gather).
// Layer1 (din=NSEG*128 -> 128) + silu -> f16 hidden in LDS; layer2 (128 -> 128) + epilogue.
enum { EPI_MSG = 0, EPI_UPD = 1, EPI_PLAIN = 2, EPI_EW = 3 };

template<int NSEG, int EPI>
__global__ __launch_bounds__(128) void k_mlp(
    const _Float16* __restrict__ seg0, const _Float16* __restrict__ seg1, const _Float16* __restrict__ seg2,
    const int* __restrict__ idx0, const int* __restrict__ idx1,
    const _Float16* __restrict__ w1t, const float* __restrict__ b1,
    const _Float16* __restrict__ w2t, const float* __restrict__ b2,
    const float* __restrict__ w2f,
    int M,
    const float* __restrict__ evf, const float* __restrict__ resf,
    float* __restrict__ outf, _Float16* __restrict__ h16a, _Float16* __restrict__ h16b)
{
  constexpr int DIN = NSEG*128;
  constexpr int KS1 = DIN/32;
  __shared__ __align__(16) _Float16 hid[64][136];   // 17.4 KB, padded stride
  const int tid = threadIdx.x;
  const int wave = tid>>5, lane = tid&31;
  const int lan16 = lane&15, hg = lane>>4;
  const int mbase = wave*16;
  const int rowg = blockIdx.x*64 + mbase + lan16;   // row owned by this lane for A
  const int rowc = (rowg < M) ? rowg : 0;

  const _Float16* a0 = seg0 + (size_t)(idx0 ? idx0[rowc] : rowc)*128;
  const _Float16* a1 = (NSEG>1) ? (seg1 + (size_t)(idx1 ? idx1[rowc] : rowc)*128) : nullptr;
  const _Float16* a2 = (NSEG>2) ? (seg2 + (size_t)rowc*128) : nullptr;

  v8f acc[8];
  #pragma unroll
  for (int t=0;t<8;t++){
    #pragma unroll
    for (int i=0;i<8;i++) acc[t][i] = 0.f;
  }

  for (int ks=0; ks<KS1; ks++){
    const int kk = ks*32;
    FragA a;
    {
      int c = kk + 8*hg;                       // chunk for VGPRs 0..3 (K = c..c+7)
      const _Float16* p;
      if (NSEG==1)      p = a0 + c;
      else if (NSEG==2) p = (c<128) ? a0+c : a1+(c-128);
      else              p = (c<128) ? a0+c : ((c<256) ? a1+(c-128) : a2+(c-256));
      a.h[0] = *(const v8h*)p;
      c = kk + 16 + 8*hg;                      // chunk for VGPRs 4..7
      if (NSEG==1)      p = a0 + c;
      else if (NSEG==2) p = (c<128) ? a0+c : a1+(c-128);
      else              p = (c<128) ? a0+c : ((c<256) ? a1+(c-128) : a2+(c-256));
      a.h[1] = *(const v8h*)p;
    }
    #pragma unroll
    for (int t=0;t<8;t++){
      FragA bf;
      const _Float16* bp = w1t + (size_t)(t*16 + lan16)*DIN + kk + 8*hg;
      bf.h[0] = *(const v8h*)bp;
      bf.h[1] = *(const v8h*)(bp + 16);
      acc[t] = __builtin_amdgcn_wmma_f32_16x16x32_f16(false, a.v, false, bf.v,
                                                      (short)0, acc[t], false, false);
    }
  }

  // bias + silu -> hidden f16 tile in LDS (C layout: lane col = lan16, rows v+8*hg)
  #pragma unroll
  for (int t=0;t<8;t++){
    const int n = t*16 + lan16;
    const float bb = b1[n];
    #pragma unroll
    for (int v=0;v<8;v++){
      float x = acc[t][v] + bb;
      hid[mbase + v + 8*hg][n] = (_Float16)silu_f(x);
    }
  }
  __syncthreads();

  if constexpr (EPI == EPI_EW) {
    // layer2 is dout=1: per-row dot with fp32 w2 + sigmoid
    const int r = tid>>1, half = tid&1;
    const int rg = blockIdx.x*64 + r;
    float s = 0.f;
    #pragma unroll 8
    for (int k=half*64; k<half*64+64; k++) s += (float)hid[r][k] * w2f[k];
    s += __shfl_xor(s, 1);
    if (half==0 && rg < M) outf[rg] = sigm_f(s + b2[0]);
    return;
  } else {
    v8f acc2[8];
    #pragma unroll
    for (int t=0;t<8;t++){
      #pragma unroll
      for (int i=0;i<8;i++) acc2[t][i] = 0.f;
    }
    #pragma unroll
    for (int ks=0; ks<4; ks++){
      const int kk = ks*32;
      FragA a;
      a.h[0] = *(const v8h*)&hid[mbase + lan16][kk + 8*hg];
      a.h[1] = *(const v8h*)&hid[mbase + lan16][kk + 16 + 8*hg];
      #pragma unroll
      for (int t=0;t<8;t++){
        FragA bf;
        const _Float16* bp = w2t + (size_t)(t*16 + lan16)*128 + kk + 8*hg;
        bf.h[0] = *(const v8h*)bp;
        bf.h[1] = *(const v8h*)(bp + 16);
        acc2[t] = __builtin_amdgcn_wmma_f32_16x16x32_f16(false, a.v, false, bf.v,
                                                         (short)0, acc2[t], false, false);
      }
    }
    #pragma unroll
    for (int t=0;t<8;t++){
      const int n = t*16 + lan16;
      const float bb = b2[n];
      #pragma unroll
      for (int v=0;v<8;v++){
        const int rl = mbase + v + 8*hg;
        const int rg2 = blockIdx.x*64 + rl;
        if (rg2 < M){
          const size_t o = (size_t)rg2*HD + n;
          const float val = acc2[t][v] + bb;
          if constexpr (EPI == EPI_MSG){
            const float e0 = evf[o];
            outf[o] = val;                          // msg (f32, for scatter)
            h16a[o] = (_Float16)(e0 + val);         // e_new  (edge_weight input)
            h16b[o] = (_Float16)(2.f*e0 + val);     // ev residual (edge_update input)
          } else if constexpr (EPI == EPI_UPD){
            outf[o] = resf[o] + val;                // hv + h_new
          } else {
            outf[o] = val;
          }
        }
      }
    }
  }
}

// ---------------- host ----------------

extern "C" void kernel_launch(void* const* d_in, const int* in_sizes, int n_in,
                              void* d_out, int out_size, void* d_ws, size_t ws_size,
                              hipStream_t stream)
{
  (void)in_sizes; (void)n_in; (void)out_size; (void)ws_size;
  const int* h_idx  = (const int*)d_in[0];
  const int* e_idx  = (const int*)d_in[1];
  const int* indeg  = (const int*)d_in[2];
  const int* outdeg = (const int*)d_in[3];
  const int* sp     = (const int*)d_in[4];
  const int* src    = (const int*)d_in[5];
  const int* dst    = (const int*)d_in[6];
  // params flattened in jax pytree (sorted-key) order
  const float* edge_norm_b = (const float*)d_in[7];
  const float* edge_norm_g = (const float*)d_in[8];
  const float* eupd_l1b = (const float*)d_in[9];
  const float* eupd_l1w = (const float*)d_in[10];
  const float* eupd_l2b = (const float*)d_in[11];
  const float* eupd_l2w = (const float*)d_in[12];
  const float* ew_l1b   = (const float*)d_in[13];
  const float* ew_l1w   = (const float*)d_in[14];
  const float* ew_l2b   = (const float*)d_in[15];
  const float* ew_l2w   = (const float*)d_in[16];
  const float* emb_e    = (const float*)d_in[17];
  const float* emb_h    = (const float*)d_in[18];
  const float* emb_in   = (const float*)d_in[19];
  const float* emb_out  = (const float*)d_in[20];
  const float* emb_sp   = (const float*)d_in[21];
  const float *msg1b[3], *msg1w[3], *msg2b[3], *msg2w[3];
  const float *upd1b[3], *upd1w[3], *upd2b[3], *upd2w[3];
  for (int l=0;l<3;l++){
    int b0 = 22 + l*8;
    msg1b[l]=(const float*)d_in[b0+0]; msg1w[l]=(const float*)d_in[b0+1];
    msg2b[l]=(const float*)d_in[b0+2]; msg2w[l]=(const float*)d_in[b0+3];
    upd1b[l]=(const float*)d_in[b0+4]; upd1w[l]=(const float*)d_in[b0+5];
    upd2b[l]=(const float*)d_in[b0+6]; upd2w[l]=(const float*)d_in[b0+7];
  }
  const float* node_norm_b = (const float*)d_in[46];
  const float* node_norm_g = (const float*)d_in[47];
  const float* out_l1b = (const float*)d_in[48];
  const float* out_l1w = (const float*)d_in[49];
  const float* out_l2b = (const float*)d_in[50];
  const float* out_l2w = (const float*)d_in[51];

  // workspace carve-out
  char* wp = (char*)d_ws;
  auto alloc = [&](size_t bytes)->void*{ void* r=(void*)wp; wp += (bytes+255)&~(size_t)255; return r; };
  float*    ev      = (float*)   alloc((size_t)EE*HD*4);
  float*    msg     = (float*)   alloc((size_t)EE*HD*4);  // reused as pre-LN edge buffer
  _Float16* ev16    = (_Float16*)alloc((size_t)EE*HD*2);
  _Float16* pos16   = (_Float16*)alloc((size_t)EE*HD*2);
  _Float16* enew16  = (_Float16*)alloc((size_t)EE*HD*2);
  _Float16* evres16 = (_Float16*)alloc((size_t)EE*HD*2);
  float*    wbuf    = (float*)   alloc((size_t)EE*4);
  float*    hv      = (float*)   alloc((size_t)NN*HD*4);
  float*    hres    = (float*)   alloc((size_t)NN*HD*4);
  float*    agg     = (float*)   alloc((size_t)NN*HD*4);
  _Float16* hv16    = (_Float16*)alloc((size_t)NN*HD*2);
  _Float16* aggn16  = (_Float16*)alloc((size_t)NN*HD*2);
  float*    deg     = (float*)   alloc((size_t)NN*4);
  _Float16 *wt_msg1[3], *wt_msg2[3], *wt_upd1[3], *wt_upd2[3];
  for (int l=0;l<3;l++){
    wt_msg1[l] = (_Float16*)alloc((size_t)384*128*2);
    wt_msg2[l] = (_Float16*)alloc((size_t)128*128*2);
    wt_upd1[l] = (_Float16*)alloc((size_t)256*128*2);
    wt_upd2[l] = (_Float16*)alloc((size_t)128*128*2);
  }
  _Float16* wt_ew1   = (_Float16*)alloc((size_t)128*128*2);
  _Float16* wt_eupd1 = (_Float16*)alloc((size_t)256*128*2);
  _Float16* wt_eupd2 = (_Float16*)alloc((size_t)128*128*2);
  _Float16* wt_out1  = (_Float16*)alloc((size_t)128*128*2);
  _Float16* wt_out2  = (_Float16*)alloc((size_t)128*128*2);

  auto gw = [&](const float* w, _Float16* wt, int din, int dout){
    int n = din*dout;
    k_wt_f16<<<(n+255)/256, 256, 0, stream>>>(w, wt, din, dout);
  };
  for (int l=0;l<3;l++){
    gw(msg1w[l], wt_msg1[l], 384,128);
    gw(msg2w[l], wt_msg2[l], 128,128);
    gw(upd1w[l], wt_upd1[l], 256,128);
    gw(upd2w[l], wt_upd2[l], 128,128);
  }
  gw(ew_l1w,   wt_ew1,   128,128);
  gw(eupd_l1w, wt_eupd1, 256,128);
  gw(eupd_l2w, wt_eupd2, 128,128);
  gw(out_l1w,  wt_out1,  128,128);
  gw(out_l2w,  wt_out2,  128,128);

  // embeddings + initial edge LN + degree
  k_node_init<<<(NN*HD+255)/256,256,0,stream>>>(h_idx, indeg, outdeg, emb_h, emb_in, emb_out, hv, hv16);
  k_edge_pre<<<(EE*HD+255)/256,256,0,stream>>>(e_idx, sp, emb_e, emb_sp, msg, pos16);
  k_ln<<<EE,128,0,stream>>>(msg, edge_norm_g, edge_norm_b, ev, ev16);
  k_fill0<<<(NN+255)/256,256,0,stream>>>(deg, NN);
  k_deg<<<(EE+255)/256,256,0,stream>>>(dst, deg);

  const int gbE = (EE+63)/64, gbN = (NN+63)/64;
  for (int l=0;l<3;l++){
    // msg MLP: A = [hv[src], hv[dst], ev]  (gather fused)
    k_mlp<3,EPI_MSG><<<gbE,128,0,stream>>>(hv16, hv16, ev16, src, dst,
        wt_msg1[l], msg1b[l], wt_msg2[l], msg2b[l], nullptr, EE,
        ev, nullptr, msg, enew16, evres16);
    // edge_weight MLP on e_new -> sigmoid scalar per edge
    k_mlp<1,EPI_EW><<<gbE,128,0,stream>>>(enew16, nullptr, nullptr, nullptr, nullptr,
        wt_ew1, ew_l1b, nullptr, ew_l2b, ew_l2w, EE,
        nullptr, nullptr, wbuf, nullptr, nullptr);
    // aggregate msg*w by dst, normalize by degree
    k_fill0<<<(NN*HD+255)/256,256,0,stream>>>(agg, NN*HD);
    k_scatter<<<EE,128,0,stream>>>(msg, wbuf, dst, agg);
    k_aggn<<<(NN*HD+255)/256,256,0,stream>>>(agg, deg, aggn16);
    // node update MLP: A = [hv, agg/deg]; residual add
    k_mlp<2,EPI_UPD><<<gbN,128,0,stream>>>(hv16, aggn16, nullptr, nullptr, nullptr,
        wt_upd1[l], upd1b[l], wt_upd2[l], upd2b[l], nullptr, NN,
        nullptr, hv, hres, nullptr, nullptr);
    k_ln<<<NN,128,0,stream>>>(hres, node_norm_g, node_norm_b, hv, hv16);
    // edge update MLP: A = [2*ev+msg, pos]
    k_mlp<2,EPI_PLAIN><<<gbE,128,0,stream>>>(evres16, pos16, nullptr, nullptr, nullptr,
        wt_eupd1, eupd_l1b, wt_eupd2, eupd_l2b, nullptr, EE,
        nullptr, nullptr, msg, nullptr, nullptr);
    k_ln<<<EE,128,0,stream>>>(msg, edge_norm_g, edge_norm_b, ev, ev16);
  }
  // output MLP
  k_mlp<1,EPI_PLAIN><<<gbN,128,0,stream>>>(hv16, nullptr, nullptr, nullptr, nullptr,
      wt_out1, out_l1b, wt_out2, out_l2b, nullptr, NN,
      nullptr, nullptr, (float*)d_out, nullptr, nullptr);
}
